// KittyCatConv_33243046871202
// MI455X (gfx1250) — matrix operand order repaired and checked by reading, not compiled, not run on hardware
//
#include <hip/hip_runtime.h>
#include <stdint.h>

typedef __attribute__((ext_vector_type(8)))  float        v8f;
typedef __attribute__((ext_vector_type(16))) __bf16       v16bf;
typedef __attribute__((ext_vector_type(4)))  unsigned int u32x4;

union FragBF { v16bf v; u32x4 q[2]; };

__device__ __forceinline__ unsigned short f2bf(float f) {
  unsigned int u = __float_as_uint(f);
  u += 0x7fffu + ((u >> 16) & 1u);
  return (unsigned short)(u >> 16);
}

// ---------------- generic fp32 -> bf16 convert ----------------
__global__ void kcvt_bf16(const float* __restrict__ in, unsigned short* __restrict__ out, int n) {
  int i = (blockIdx.x * 256 + threadIdx.x) * 4;
#pragma unroll
  for (int k = 0; k < 4; ++k) {
    int j = i + k;
    if (j < n) out[j] = f2bf(in[j]);
  }
}

// ---------------- conv1d as WMMA GEMM ----------------
// Y[b,co,t] = sum_tap sum_ci W[co,ci,tap] * X[b,ci,t+tap-p]   (bias omitted: cancels in BN)
// Block tile: 64 (co) x 128 (t). 8 waves = 2(m) x 4(n), each wave 2x2 WMMA 16x16 tiles.
__global__ __launch_bounds__(256) void kconv_wmma(
    const unsigned short* __restrict__ W,   // (512,512,f) bf16 bits
    const unsigned short* __restrict__ X,   // (4,512,2048) bf16 bits
    float* __restrict__ Y, int f) {
  __shared__ __align__(16) unsigned short As[64][40];
  __shared__ __align__(16) unsigned short Bs[128][40];
  const int p = (f - 1) >> 1;
  const int tid = threadIdx.x;
  const int t0  = blockIdx.x * 128;
  const int co0 = blockIdx.y * 64;
  const int b   = blockIdx.z;
  const unsigned short* Xb = X + (size_t)b * (512 * 2048);
  const int wave = tid >> 5, lane = tid & 31;
  const int wm = wave >> 2, wn = wave & 3;
  const int mrow = lane & 15, khalf = lane >> 4;
  v8f acc[2][2];
#pragma unroll
  for (int a = 0; a < 2; ++a)
#pragma unroll
    for (int c = 0; c < 2; ++c) acc[a][c] = (v8f){0.f,0.f,0.f,0.f,0.f,0.f,0.f,0.f};

  for (int tap = 0; tap < f; ++tap) {
    const int s = tap - p;
    for (int cic = 0; cic < 16; ++cic) {
      __syncthreads();
      // A tile: W[co0..+63, cic*32..+31, tap] -> As[row][col]
#pragma unroll
      for (int g = 0; g < 8; ++g) {
        int e = g * 256 + tid; int col = e & 31; int row = e >> 5;
        As[row][col] = W[((size_t)(co0 + row) * 512 + (size_t)(cic * 32 + col)) * f + tap];
      }
      // B tile (transposed): Bs[t_local][ci_local] = X[b, cic*32+ci, t0+t_local+s]
#pragma unroll
      for (int g = 0; g < 16; ++g) {
        int e = g * 256 + tid; int tl = e & 127; int ci = e >> 7;
        int t = t0 + tl + s;
        unsigned short v = 0;
        if (t >= 0 && t < 2048) v = Xb[(size_t)(cic * 32 + ci) * 2048 + t];
        Bs[tl][ci] = v;
      }
      __syncthreads();
      FragBF afr[2], bfr[2];
#pragma unroll
      for (int tm = 0; tm < 2; ++tm) {   // A: element e -> K = (e>>3)*16 + khalf*8 + (e&7)
        int row = wm * 32 + tm * 16 + mrow;
        afr[tm].q[0] = *(const u32x4*)&As[row][khalf * 8];
        afr[tm].q[1] = *(const u32x4*)&As[row][16 + khalf * 8];
      }
#pragma unroll
      for (int tn = 0; tn < 2; ++tn) {   // B: element e -> K = (e>>3)*16 + khalf*8 + (e&7)
        int row = wn * 32 + tn * 16 + mrow;
        bfr[tn].q[0] = *(const u32x4*)&Bs[row][khalf * 8];
        bfr[tn].q[1] = *(const u32x4*)&Bs[row][16 + khalf * 8];
      }
#pragma unroll
      for (int tm = 0; tm < 2; ++tm)
#pragma unroll
        for (int tn = 0; tn < 2; ++tn)
          acc[tm][tn] = __builtin_amdgcn_wmma_f32_16x16x32_bf16(
              false, afr[tm].v, false, bfr[tn].v, (short)0, acc[tm][tn], false, false);
    }
  }
  const int nn = lane & 15, mh = lane >> 4;
#pragma unroll
  for (int tm = 0; tm < 2; ++tm)
#pragma unroll
    for (int tn = 0; tn < 2; ++tn)
#pragma unroll
      for (int r = 0; r < 8; ++r) {
        int m  = r + mh * 8;
        int co = co0 + wm * 32 + tm * 16 + m;
        int t  = t0 + wn * 32 + tn * 16 + nn;
        Y[(size_t)b * 1048576 + (size_t)co * 2048 + t] = acc[tm][tn][r];
      }
}

// ---------------- BN stats (biased var over (B,L)) ----------------
__global__ __launch_bounds__(256) void kbn_stats(const float* __restrict__ Y,
                                                 float* __restrict__ mean,
                                                 float* __restrict__ istd) {
  __shared__ float s1[256], s2[256];
  int c = blockIdx.x, tid = threadIdx.x;
  float a = 0.f, b = 0.f;
  for (int e = tid; e < 8192; e += 256) {
    int bb = e >> 11, t = e & 2047;
    float x = Y[(size_t)bb * 1048576 + (size_t)c * 2048 + t];
    a += x; b += x * x;
  }
  s1[tid] = a; s2[tid] = b; __syncthreads();
  for (int st = 128; st > 0; st >>= 1) {
    if (tid < st) { s1[tid] += s1[tid + st]; s2[tid] += s2[tid + st]; }
    __syncthreads();
  }
  if (tid == 0) {
    float m = s1[0] * (1.f / 8192.f);
    float v = s2[0] * (1.f / 8192.f) - m * m;
    if (v < 0.f) v = 0.f;
    mean[c] = m;
    istd[c] = rsqrtf(v + 1e-5f);
  }
}

// ---------------- BN apply + ELU -> bf16 state ----------------
__global__ void kbn_apply_state(const float* __restrict__ Y, const float* __restrict__ mean,
                                const float* __restrict__ istd, const float* __restrict__ g,
                                const float* __restrict__ beta, unsigned short* __restrict__ st) {
  int i = (blockIdx.x * 256 + threadIdx.x) * 4;
#pragma unroll
  for (int k = 0; k < 4; ++k) {
    int e = i + k;
    int c = (e >> 11) & 511;
    float x = Y[e];
    float v = g[c] * (x - mean[c]) * istd[c] + beta[c];
    v = v > 0.f ? v : (__expf(v) - 1.f);
    st[e] = f2bf(v);
  }
}

// ---------------- BN apply + ELU + proj-dot (groups of 64 along t) ----------------
__global__ __launch_bounds__(256) void kbn_apply_score(
    const float* __restrict__ Y, const float* __restrict__ mean, const float* __restrict__ istd,
    const float* __restrict__ g, const float* __restrict__ beta,
    const float* __restrict__ pv, float* __restrict__ score, int fidx) {
  __shared__ float wsum[8];
  int tid = threadIdx.x;
  int gid = blockIdx.x * 4 + (tid >> 6);
  int d = tid & 63;
  int e = gid * 64 + d;
  int c = (e >> 11) & 511;
  float x = Y[e];
  float v = g[c] * (x - mean[c]) * istd[c] + beta[c];
  v = v > 0.f ? v : (__expf(v) - 1.f);
  float prod = v * pv[d];
#pragma unroll
  for (int m = 16; m > 0; m >>= 1) prod += __shfl_xor(prod, m, 32);
  if ((tid & 31) == 0) wsum[tid >> 5] = prod;
  __syncthreads();
  if ((tid & 63) == 0) score[fidx * 65536 + gid] = wsum[tid >> 5] + wsum[(tid >> 5) + 1];
}

// ---------------- per-(b,h) bitonic sort desc, keep top 2048 of 6144 ----------------
__global__ __launch_bounds__(256) void ksort_top(const float* __restrict__ score,
                                                 float* __restrict__ qtop) {
  __shared__ float s[8192];
  int bh = blockIdx.x, tid = threadIdx.x;
  for (int i = tid; i < 8192; i += 256)
    s[i] = (i < 6144) ? score[(size_t)bh * 6144 + i] : -3.402823e38f;
  __syncthreads();
  for (int k = 2; k <= 8192; k <<= 1) {
    for (int j = k >> 1; j > 0; j >>= 1) {
      for (int i = tid; i < 8192; i += 256) {
        int ixj = i ^ j;
        if (ixj > i) {
          float a = s[i], b = s[ixj];
          bool desc = ((i & k) == 0);
          if (desc ? (a < b) : (a > b)) { s[i] = b; s[ixj] = a; }
        }
      }
      __syncthreads();
    }
  }
  for (int i = tid; i < 2048; i += 256) qtop[(size_t)bh * 2048 + i] = s[i];
}

// ---------------- k_mean over the 3 filter blocks ----------------
__global__ void kkmean(const float* __restrict__ ks, float* __restrict__ km) {
  int u = blockIdx.x * 256 + threadIdx.x;   // 65536 total
  int bh = u >> 11, t = u & 2047;
  const float* p = ks + (size_t)bh * 6144;
  km[u] = (p[t] + p[2048 + t] + p[4096 + t]) * (1.f / 3.f);
}

// ---------------- fused softmax + attn store + context WMMA ----------------
// attn[b,h,q,j] = softmax_j(c0 * q_top[q] * k_mean[j]);  context = attn @ V
__global__ __launch_bounds__(256) void kattn_ctx(
    const float* __restrict__ qtop, const float* __restrict__ km,
    const float* __restrict__ pbq, const float* __restrict__ pbk,
    const unsigned short* __restrict__ Vbf,
    float* __restrict__ ctx, float* __restrict__ attn) {
  __shared__ float km_s[2048];
  __shared__ __align__(16) unsigned short Vt[64][40];
  __shared__ float red[256];
  __shared__ float rowS[8][16], rowM[8][16], rowD[8][16];
  __shared__ float sc[3];   // c0, kmax, kmin
  const int tid = threadIdx.x;
  const int bh = blockIdx.y;
  const int q0 = blockIdx.x * 128;
  const int wave = tid >> 5, lane = tid & 31;

  for (int j = tid; j < 2048; j += 256) km_s[j] = km[(size_t)bh * 2048 + j];
  __syncthreads();
  float pmax = -3.402823e38f, pmin = 3.402823e38f;
  for (int j = tid; j < 2048; j += 256) {
    float v = km_s[j]; pmax = fmaxf(pmax, v); pmin = fminf(pmin, v);
  }
  red[tid] = pmax; __syncthreads();
  for (int st = 128; st > 0; st >>= 1) { if (tid < st) red[tid] = fmaxf(red[tid], red[tid + st]); __syncthreads(); }
  if (tid == 0) sc[1] = red[0];
  __syncthreads();
  red[tid] = pmin; __syncthreads();
  for (int st = 128; st > 0; st >>= 1) { if (tid < st) red[tid] = fminf(red[tid], red[tid + st]); __syncthreads(); }
  if (tid == 0) {
    sc[2] = red[0];
    float c0 = 0.f;
    for (int d = 0; d < 64; ++d) c0 += pbq[d] * pbk[d];
    sc[0] = c0 * 0.125f;   // 1/sqrt(64)
  }
  __syncthreads();
  const float c0 = sc[0], kmax = sc[1], kmin = sc[2];

  const int qw = q0 + wave * 16;
  for (int r = 0; r < 16; ++r) {
    float s  = c0 * qtop[(size_t)bh * 2048 + qw + r];
    float rm = (s >= 0.f) ? s * kmax : s * kmin;
    float part = 0.f;
    for (int j = lane; j < 2048; j += 32) part += __expf(s * km_s[j] - rm);
#pragma unroll
    for (int m = 16; m > 0; m >>= 1) part += __shfl_xor(part, m, 32);
    float invd = 1.f / part;
    if (lane == 0) { rowS[wave][r] = s; rowM[wave][r] = rm; rowD[wave][r] = invd; }
    float* arow = attn + ((size_t)bh * 2048 + (size_t)(qw + r)) * 2048;
    for (int j0 = 0; j0 < 2048; j0 += 128) {
      int j = j0 + lane * 4;
      float4 o;
      o.x = __expf(s * km_s[j + 0] - rm) * invd;
      o.y = __expf(s * km_s[j + 1] - rm) * invd;
      o.z = __expf(s * km_s[j + 2] - rm) * invd;
      o.w = __expf(s * km_s[j + 3] - rm) * invd;
      *(float4*)(arow + j) = o;
    }
  }
  __syncthreads();

  const int mrow = lane & 15, khalf = lane >> 4;
  const float sR = rowS[wave][mrow], mR = rowM[wave][mrow], dR = rowD[wave][mrow];
  v8f acc[4];
#pragma unroll
  for (int nt = 0; nt < 4; ++nt) acc[nt] = (v8f){0.f,0.f,0.f,0.f,0.f,0.f,0.f,0.f};

  for (int kt = 0; kt < 64; ++kt) {
    __syncthreads();
#pragma unroll
    for (int g = 0; g < 8; ++g) {        // stage V[kt*32..+31][0..63] transposed
      int e = g * 256 + tid; int d = e & 63; int kk = e >> 6;
      Vt[d][kk] = Vbf[((size_t)bh * 2048 + (size_t)(kt * 32 + kk)) * 64 + d];
    }
    __syncthreads();
    FragBF af;
    unsigned int wrd[8];
#pragma unroll
    for (int i = 0; i < 8; ++i) {        // A element e -> K = (e>>3)*16 + khalf*8 + (e&7)
      int e0 = 2 * i, e1 = 2 * i + 1;
      int k0 = (e0 >> 3) * 16 + khalf * 8 + (e0 & 7);
      int k1 = (e1 >> 3) * 16 + khalf * 8 + (e1 & 7);
      unsigned int lo = f2bf(__expf(sR * km_s[kt * 32 + k0] - mR) * dR);
      unsigned int hi = f2bf(__expf(sR * km_s[kt * 32 + k1] - mR) * dR);
      wrd[i] = lo | (hi << 16);
    }
    af.q[0] = (u32x4){wrd[0], wrd[1], wrd[2], wrd[3]};
    af.q[1] = (u32x4){wrd[4], wrd[5], wrd[6], wrd[7]};
#pragma unroll
    for (int nt = 0; nt < 4; ++nt) {
      FragBF bf;
      int rowv = nt * 16 + mrow;
      bf.q[0] = *(const u32x4*)&Vt[rowv][khalf * 8];
      bf.q[1] = *(const u32x4*)&Vt[rowv][16 + khalf * 8];
      acc[nt] = __builtin_amdgcn_wmma_f32_16x16x32_bf16(
          false, af.v, false, bf.v, (short)0, acc[nt], false, false);
    }
  }
  const int nn = lane & 15, mh = lane >> 4;
#pragma unroll
  for (int nt = 0; nt < 4; ++nt)
#pragma unroll
    for (int r = 0; r < 8; ++r) {
      int m = r + mh * 8;
      ctx[((size_t)bh * 2048 + (size_t)(qw + m)) * 64 + nt * 16 + nn] = acc[nt][r];
    }
}

// ======================== host launcher ========================
extern "C" void kernel_launch(void* const* d_in, const int* in_sizes, int n_in,
                              void* d_out, int out_size, void* d_ws, size_t ws_size,
                              hipStream_t stream) {
  (void)in_sizes; (void)n_in; (void)out_size; (void)ws_size;
  const float* Q    = (const float*)d_in[0];
  const float* K    = (const float*)d_in[1];
  const float* V    = (const float*)d_in[2];
  const float* bn_g = (const float*)d_in[16];
  const float* bn_b = (const float*)d_in[17];
  const float* pq   = (const float*)d_in[18];
  const float* pk   = (const float*)d_in[19];
  const float* pbq  = (const float*)d_in[20];
  const float* pbk  = (const float*)d_in[21];
  const float* wq_f32[3] = { (const float*)d_in[4],  (const float*)d_in[8],  (const float*)d_in[12] };
  const float* wk_f32[3] = { (const float*)d_in[6],  (const float*)d_in[10], (const float*)d_in[14] };

  const int    F[3]    = {1, 3, 9};
  const size_t woff[3] = {0, 262144, 1048576};
  const int    wlen[3] = {262144, 786432, 2359296};
  const int    NACT = 4194304;   // 4*512*2048

  char* w = (char*)d_ws;
  auto alloc = [&](size_t bytes) { char* r = w; w += (bytes + 255) & ~(size_t)255; return r; };
  unsigned short* wq_bf = (unsigned short*)alloc((size_t)3407872 * 2);
  unsigned short* wk_bf = (unsigned short*)alloc((size_t)3407872 * 2);
  unsigned short* st_q  = (unsigned short*)alloc((size_t)NACT * 2);
  unsigned short* st_k  = (unsigned short*)alloc((size_t)NACT * 2);
  float*          Ybuf  = (float*)alloc((size_t)NACT * 4);
  float*          meanb = (float*)alloc(512 * 4);
  float*          istdb = (float*)alloc(512 * 4);
  float*          qsc   = (float*)alloc((size_t)196608 * 4);
  float*          ksc   = (float*)alloc((size_t)196608 * 4);
  float*          qtop  = (float*)alloc((size_t)65536 * 4);
  float*          kmn   = (float*)alloc((size_t)65536 * 4);
  unsigned short* vbf   = (unsigned short*)alloc((size_t)NACT * 2);

  auto cvt = [&](const float* src, unsigned short* dst, int n) {
    kcvt_bf16<<<(n + 1023) / 1024, 256, 0, stream>>>(src, dst, n);
  };
  for (int i = 0; i < 3; ++i) {
    cvt(wq_f32[i], wq_bf + woff[i], wlen[i]);
    cvt(wk_f32[i], wk_bf + woff[i], wlen[i]);
  }
  cvt(Q, st_q, NACT);
  cvt(K, st_k, NACT);
  cvt(V, vbf, NACT);

  const dim3 gConv(16, 8, 4);
  for (int s = 0; s < 2; ++s) {
    unsigned short* wbf   = s ? wk_bf : wq_bf;
    unsigned short* state = s ? st_k  : st_q;
    float*          score = s ? ksc   : qsc;
    const float*    pv    = s ? pk    : pq;
    for (int i = 0; i < 3; ++i) {
      // state <- elu(bn(conv_i(state)))
      kconv_wmma<<<gConv, 256, 0, stream>>>(wbf + woff[i], state, Ybuf, F[i]);
      kbn_stats<<<512, 256, 0, stream>>>(Ybuf, meanb, istdb);
      kbn_apply_state<<<4096, 256, 0, stream>>>(Ybuf, meanb, istdb, bn_g, bn_b, state);
      // score block i <- proj(elu(bn(conv_i(state))))
      kconv_wmma<<<gConv, 256, 0, stream>>>(wbf + woff[i], state, Ybuf, F[i]);
      kbn_stats<<<512, 256, 0, stream>>>(Ybuf, meanb, istdb);
      kbn_apply_score<<<16384, 256, 0, stream>>>(Ybuf, meanb, istdb, bn_g, bn_b, pv, score, i);
    }
  }

  ksort_top<<<32, 256, 0, stream>>>(qsc, qtop);
  kkmean<<<256, 256, 0, stream>>>(ksc, kmn);

  float* ctx  = (float*)d_out;                 // (4,8,2048,64)
  float* attn = (float*)d_out + 4194304;       // (4,8,2048,2048)
  kattn_ctx<<<dim3(16, 32), 256, 0, stream>>>(qtop, kmn, pbq, pbk, vbf, ctx, attn);
}
